// PointnetFPModule_85066122265454
// MI455X (gfx1250) — matrix operand order repaired and checked by reading, not compiled, not run on hardware
//
#include <hip/hip_runtime.h>
#include <hip/hip_bf16.h>

// ---------------------------------------------------------------------------
// Problem constants (from reference setup_inputs)
// ---------------------------------------------------------------------------
#define BATCH 16
#define N1    4096
#define N2    1024
#define C1    128
#define C2    256
#define CIN   384      // C1 + C2
#define CH    256
#define COUT  256
#define NTOT  (BATCH * N1)   // 65536 points total

typedef __attribute__((ext_vector_type(8)))  float        v8f;
typedef __attribute__((ext_vector_type(8)))  __bf16       v8bf;
typedef __attribute__((ext_vector_type(16))) __bf16       v16bf;
typedef __attribute__((ext_vector_type(4)))  unsigned int v4u;
typedef __attribute__((ext_vector_type(8)))  int          v8i;
typedef __attribute__((ext_vector_type(4)))  int          v4i;

#if defined(__has_builtin)
#if __has_builtin(__builtin_amdgcn_tensor_load_to_lds) && __has_builtin(__builtin_amdgcn_s_wait_tensorcnt)
#define USE_TDM 1
#endif
#endif
#ifndef USE_TDM
#define USE_TDM 0
#endif

__device__ __forceinline__ unsigned short f2bf(float f) {
    unsigned int u = __float_as_uint(f);
    unsigned int r = u + 0x7FFFu + ((u >> 16) & 1u);   // round-to-nearest-even
    return (unsigned short)(r >> 16);
}
__device__ __forceinline__ float bf2f(unsigned short h) {
    return __uint_as_float(((unsigned int)h) << 16);
}

// ---------------------------------------------------------------------------
// Kernel 0: convert W1/W2 weights fp32 -> bf16
// ---------------------------------------------------------------------------
__global__ __launch_bounds__(256)
void convert_w_kernel(const float* __restrict__ W1, const float* __restrict__ W2,
                      unsigned short* __restrict__ W1b, unsigned short* __restrict__ W2b) {
    int i = blockIdx.x * 256 + threadIdx.x;
    if (i < CH * CIN)   W1b[i] = f2bf(W1[i]);
    if (i < COUT * CH)  W2b[i] = f2bf(W2[i]);
}

// ---------------------------------------------------------------------------
// Kernel 1: 3-NN inverse-distance interpolation + concat, bf16 output
// Output xb layout: (B, N1, 384) row-major. One wave32 per (b, n) point.
// ---------------------------------------------------------------------------
__global__ __launch_bounds__(256)
void interp_concat_kernel(const float* __restrict__ xyz1, const float* __restrict__ xyz2,
                          const float* __restrict__ points1, const float* __restrict__ points2,
                          unsigned short* __restrict__ xb) {
    const int wave = (blockIdx.x * blockDim.x + threadIdx.x) >> 5;
    const int lane = threadIdx.x & 31;
    const int b = wave >> 12;
    const int n = wave & (N1 - 1);

    const float qx = xyz1[((b * 3 + 0) << 12) + n];
    const float qy = xyz1[((b * 3 + 1) << 12) + n];
    const float qz = xyz1[((b * 3 + 2) << 12) + n];

    const float* x2 = xyz2 + (size_t)b * 3 * N2;
    float bd0 = 3.4e38f, bd1 = 3.4e38f, bd2 = 3.4e38f;
    int   bi0 = -1, bi1 = -1, bi2 = -1;
    #pragma unroll 4
    for (int it = 0; it < N2 / 32; ++it) {
        int j = it * 32 + lane;
        float dx = x2[j] - qx;
        float dy = x2[N2 + j] - qy;
        float dz = x2[2 * N2 + j] - qz;
        float d = dx * dx + dy * dy + dz * dz;
        if (d < bd2) {
            if (d < bd0)      { bd2 = bd1; bi2 = bi1; bd1 = bd0; bi1 = bi0; bd0 = d; bi0 = j; }
            else if (d < bd1) { bd2 = bd1; bi2 = bi1; bd1 = d;  bi1 = j; }
            else              { bd2 = d;  bi2 = j; }
        }
    }

    // merge across lanes: 3 rounds of argmin (ties -> lower index, as jax top_k)
    int p = 0;
    float seld0, seld1, seld2; int seli0, seli1, seli2;
    #pragma unroll
    for (int k = 0; k < 3; ++k) {
        float d; int id;
        if      (p == 0) { d = bd0; id = bi0; }
        else if (p == 1) { d = bd1; id = bi1; }
        else if (p == 2) { d = bd2; id = bi2; }
        else             { d = 3.4e38f; id = 0x7fffffff; }
        const int sub = id;
        #pragma unroll
        for (int off = 16; off >= 1; off >>= 1) {
            float d2 = __shfl_xor(d, off, 32);
            int   i2 = __shfl_xor(id, off, 32);
            if (d2 < d || (d2 == d && i2 < id)) { d = d2; id = i2; }
        }
        if (k == 0) { seld0 = d; seli0 = id; }
        if (k == 1) { seld1 = d; seli1 = id; }
        if (k == 2) { seld2 = d; seli2 = id; }
        if (sub == id) ++p;
    }

    float w0 = 1.f / fmaxf(seld0, 1e-10f);
    float w1 = 1.f / fmaxf(seld1, 1e-10f);
    float w2 = 1.f / fmaxf(seld2, 1e-10f);
    float ws = w0 + w1 + w2;
    w0 /= ws; w1 /= ws; w2 /= ws;

    unsigned short* xrow = xb + (size_t)((b << 12) + n) * CIN;
    const float* p2 = points2 + (size_t)b * C2 * N2;
    #pragma unroll
    for (int cc = 0; cc < C2 / 32; ++cc) {
        int c = cc * 32 + lane;
        const float* pc = p2 + (size_t)c * N2;
        float v = w0 * pc[seli0] + w1 * pc[seli1] + w2 * pc[seli2];
        xrow[C1 + c] = f2bf(v);
    }
    const float* p1 = points1 + (size_t)b * C1 * N1;
    #pragma unroll
    for (int cc = 0; cc < C1 / 32; ++cc) {
        int c = cc * 32 + lane;
        xrow[c] = f2bf(p1[(size_t)c * N1 + n]);
    }
}

// ---------------------------------------------------------------------------
// LDS-staged GEMM via v_wmma_f32_16x16x32_bf16.
//   Block = 256 threads = 8 waves. Block owns a 64-point N strip and the FULL
//   M=256: the B tile (64 x KDIM bf16) is staged into LDS ONCE (TDM if
//   available, cooperative copy otherwise) and reused by all 16 M-tiles.
//   Wave w computes M-tiles {2w, 2w+1}: B fragments from LDS feed 2 WMMAs.
//
// LDS image is padded: +4 bf16 per 128 elements (matches TDM pad_interval=256B,
// pad_amount=8B) so per-lane ds_load_b128 of rows 528B/792B apart avoids the
// 16-way bank conflict of the natural 768B stride.
//   element e (linear, rows contiguous) -> lds elem (e>>7)*132 + (e&127)
// ---------------------------------------------------------------------------
template<int KDIM, bool F32OUT>
__global__ __launch_bounds__(256)
void gemm_kernel(const unsigned short* __restrict__ Wb,
                 const unsigned short* __restrict__ Xb,
                 unsigned short* __restrict__ Ybf,   // !F32OUT: (B*N1, 256) bf16
                 float* __restrict__ Yf,             //  F32OUT: (B, 256, N1) f32
                 float* __restrict__ sum, float* __restrict__ sumsq) {
    constexpr int SROW = (KDIM / 128) * 132;         // padded row stride (elements)
    __shared__ unsigned short Bt[64 * SROW];

    const int b  = blockIdx.x >> 6;
    const int n0 = (blockIdx.x & 63) << 6;
    const int w    = threadIdx.x >> 5;
    const int lane = threadIdx.x & 31;
    const int half = lane >> 4;
    const int lm   = lane & 15;

    const unsigned short* src = Xb + (size_t)((b << 12) + n0) * KDIM;

#if USE_TDM
    if (threadIdx.x < 32) {
        // Tensor DMA: contiguous 64*KDIM*2 bytes -> LDS with padding applied.
        const unsigned long long ga = (unsigned long long)(const void*)src;
        const unsigned int ldsa = (unsigned int)((unsigned long long)(size_t)&Bt[0]);
        const unsigned int nelem = (64u * KDIM * 2u) / 8u;   // 8-byte units
        v4u g0;
        g0[0] = 1u;                                          // count=1, user mode
        g0[1] = ldsa;                                        // lds_addr
        g0[2] = (unsigned int)(ga & 0xFFFFFFFFull);          // global_addr[31:0]
        g0[3] = (unsigned int)((ga >> 32) & 0x1FFFFFFull) | (2u << 30);  // [56:32] | type=2
        v8i g1;
        g1[0] = (3 << 16)          // data_size = 8B
              | (1 << 20)          // pad_enable
              | (5 << 22)          // pad_interval: 2^(5+3)=256 bytes
              | (1 << 25);         // pad_amount: 2 DWORDs = 8 bytes
        g1[1] = (int)((nelem & 0xFFFFu) << 16);              // tensor_dim0[15:0]
        g1[2] = (int)((nelem >> 16) & 0xFFFFu) | (1 << 16);  // tensor_dim0[31:16], tensor_dim1=1
        g1[3] = (int)((nelem & 0xFFFFu) << 16);              // tile_dim0 = nelem
        g1[4] = 1;                                           // tile_dim1 = 1
        g1[5] = (int)nelem;                                  // tensor_dim0_stride
        g1[6] = 0;
        g1[7] = 0;
        v4i z4 = {0, 0, 0, 0};
        v8i z8 = {0, 0, 0, 0, 0, 0, 0, 0};
        // 6-arg form (clang-23 / therock-10.0 headers): (g0, g1, g2, g3, g4, cpol)
        __builtin_amdgcn_tensor_load_to_lds(g0, g1, z4, z4, z8, 0);
        __builtin_amdgcn_s_wait_tensorcnt(0);
    }
#else
    {
        constexpr int CHUNKS = (64 * KDIM * 2) / 16;   // 16B chunks
        const uint4* src4 = (const uint4*)src;
        for (int i = threadIdx.x; i < CHUNKS; i += 256) {
            const int e = i * 8;                              // element index
            const int loff = ((e >> 7) * 132) + (e & 127);    // padded LDS offset
            *(uint4*)(&Bt[loff]) = src4[i];
        }
    }
#endif
    __syncthreads();

    const int akoff = half ? 16 : 0;
    const int bkoff = half ? 16 : 0;
    const unsigned short* arow0 = Wb + (size_t)(w * 32 + 0  + lm) * KDIM;
    const unsigned short* arow1 = Wb + (size_t)(w * 32 + 16 + lm) * KDIM;

    v8f acc[2][4];
    #pragma unroll
    for (int mt = 0; mt < 2; ++mt)
        #pragma unroll
        for (int nn = 0; nn < 4; ++nn) acc[mt][nn] = (v8f){};

    #pragma unroll 2
    for (int k0 = 0; k0 < KDIM; k0 += 32) {
        v8bf a0lo = *(const v8bf*)(arow0 + k0 + akoff);
        v8bf a0hi = *(const v8bf*)(arow0 + k0 + akoff + 8);
        v8bf a1lo = *(const v8bf*)(arow1 + k0 + akoff);
        v8bf a1hi = *(const v8bf*)(arow1 + k0 + akoff + 8);
        v16bf a0 = __builtin_shufflevector(a0lo, a0hi, 0,1,2,3,4,5,6,7,8,9,10,11,12,13,14,15);
        v16bf a1 = __builtin_shufflevector(a1lo, a1hi, 0,1,2,3,4,5,6,7,8,9,10,11,12,13,14,15);
        const int eoff = k0 + bkoff;                          // multiple of 16
        const int koff = ((eoff >> 7) * 132) + (eoff & 127);  // padded
        #pragma unroll
        for (int nn = 0; nn < 4; ++nn) {
            const int row = nn * 16 + lm;
            const unsigned short* bp = &Bt[row * SROW + koff];
            v8bf blo = *(const v8bf*)(bp);
            v8bf bhi = *(const v8bf*)(bp + 8);
            v16bf bm = __builtin_shufflevector(blo, bhi, 0,1,2,3,4,5,6,7,8,9,10,11,12,13,14,15);
            acc[0][nn] = __builtin_amdgcn_wmma_f32_16x16x32_bf16(
                false, a0, false, bm, (short)0, acc[0][nn], false, false);
            acc[1][nn] = __builtin_amdgcn_wmma_f32_16x16x32_bf16(
                false, a1, false, bm, (short)0, acc[1][nn], false, false);
        }
    }

    // epilogue: per-channel stats + store
    #pragma unroll
    for (int mt = 0; mt < 2; ++mt) {
        const int m0 = w * 32 + mt * 16;
        float s[8], q[8];
        #pragma unroll
        for (int r = 0; r < 8; ++r) { s[r] = 0.f; q[r] = 0.f; }

        #pragma unroll
        for (int nn = 0; nn < 4; ++nn) {
            const v8f& c = acc[mt][nn];
            #pragma unroll
            for (int r = 0; r < 8; ++r) { float v = c[r]; s[r] += v; q[r] += v * v; }
            const int n = n0 + nn * 16 + lm;
            if constexpr (!F32OUT) {
                union { unsigned short u16[8]; uint4 u4; } pk;
                #pragma unroll
                for (int r = 0; r < 8; ++r) pk.u16[r] = f2bf(c[r]);
                *(uint4*)(Ybf + (size_t)((b << 12) + n) * CH + m0 + (half << 3)) = pk.u4;
            } else {
                #pragma unroll
                for (int r = 0; r < 8; ++r) {
                    int m = m0 + (half << 3) + r;
                    Yf[((size_t)((b << 8) + m) << 12) + n] = c[r];
                }
            }
        }
        // reduce over the 16 lanes of each half (same channel set per half)
        #pragma unroll
        for (int r = 0; r < 8; ++r) {
            #pragma unroll
            for (int off = 1; off <= 8; off <<= 1) {
                s[r] += __shfl_xor(s[r], off, 32);
                q[r] += __shfl_xor(q[r], off, 32);
            }
        }
        if (lm == 0) {
            #pragma unroll
            for (int r = 0; r < 8; ++r) {
                int m = m0 + (half << 3) + r;
                atomicAdd(&sum[m], s[r]);
                atomicAdd(&sumsq[m], q[r]);
            }
        }
    }
}

// ---------------------------------------------------------------------------
__global__ __launch_bounds__(256)
void bn_finalize_kernel(const float* __restrict__ sum, const float* __restrict__ sumsq,
                        const float* __restrict__ gamma, const float* __restrict__ beta,
                        float* __restrict__ scale, float* __restrict__ shift, float inv_count) {
    int c = threadIdx.x;
    float mean = sum[c] * inv_count;
    float var  = sumsq[c] * inv_count - mean * mean;
    float sc   = gamma[c] * rsqrtf(var + 1e-5f);
    scale[c] = sc;
    shift[c] = beta[c] - mean * sc;
}

__global__ __launch_bounds__(256)
void bn_apply_bf16_kernel(unsigned short* __restrict__ y,
                          const float* __restrict__ scale, const float* __restrict__ shift) {
    size_t base = ((size_t)blockIdx.x * 256 + threadIdx.x) * 8;
    int c0 = (int)(base & (CH - 1));
    union { unsigned short u16[8]; uint4 u4; } pk;
    pk.u4 = *(const uint4*)(y + base);
    #pragma unroll
    for (int r = 0; r < 8; ++r)
        pk.u16[r] = f2bf(fmaxf(bf2f(pk.u16[r]) * scale[c0 + r] + shift[c0 + r], 0.f));
    *(uint4*)(y + base) = pk.u4;
}

__global__ __launch_bounds__(256)
void bn_apply_f32_kernel(float* __restrict__ y,
                         const float* __restrict__ scale, const float* __restrict__ shift) {
    size_t i = ((size_t)blockIdx.x * 256 + threadIdx.x) * 4;
    int c = (int)((i >> 12) & (COUT - 1));
    float4 v = *(float4*)(y + i);
    float sc = scale[c], sh = shift[c];
    v.x = fmaxf(v.x * sc + sh, 0.f);
    v.y = fmaxf(v.y * sc + sh, 0.f);
    v.z = fmaxf(v.z * sc + sh, 0.f);
    v.w = fmaxf(v.w * sc + sh, 0.f);
    *(float4*)(y + i) = v;
}

// ---------------------------------------------------------------------------
extern "C" void kernel_launch(void* const* d_in, const int* in_sizes, int n_in,
                              void* d_out, int out_size, void* d_ws, size_t ws_size,
                              hipStream_t stream) {
    const float* xyz1    = (const float*)d_in[0];
    const float* xyz2    = (const float*)d_in[1];
    const float* points1 = (const float*)d_in[2];
    const float* points2 = (const float*)d_in[3];
    const float* W1      = (const float*)d_in[4];
    const float* gamma1  = (const float*)d_in[5];
    const float* beta1   = (const float*)d_in[6];
    const float* W2      = (const float*)d_in[7];
    const float* gamma2  = (const float*)d_in[8];
    const float* beta2   = (const float*)d_in[9];
    float* out = (float*)d_out;

    // workspace layout (16B-aligned); ~84.2 MB total
    char* ws = (char*)d_ws;
    unsigned short* W1b = (unsigned short*)ws;                       // 256*384*2
    unsigned short* W2b = (unsigned short*)(ws + 196608);            // 256*256*2
    unsigned short* xb  = (unsigned short*)(ws + 327680);            // 65536*384*2
    unsigned short* y1b = (unsigned short*)(ws + 50659328);          // 65536*256*2
    float* stats = (float*)(ws + 84213760);                          // 8 * 256 f32
    float* sum1 = stats +    0, *sq1 = stats +  256, *sc1 = stats +  512, *sh1 = stats +  768;
    float* sum2 = stats + 1024, *sq2 = stats + 1280, *sc2 = stats + 1536, *sh2 = stats + 1792;

    (void)hipMemsetAsync(stats, 0, 8 * 256 * sizeof(float), stream);

    convert_w_kernel<<<(CH * CIN + 255) / 256, 256, 0, stream>>>(W1, W2, W1b, W2b);

    interp_concat_kernel<<<NTOT / 8, 256, 0, stream>>>(xyz1, xyz2, points1, points2, xb);

    // GEMMs: 1024 blocks, each = full M=256 over a 64-point strip, B tile in LDS
    gemm_kernel<CIN, false><<<BATCH * (N1 / 64), 256, 0, stream>>>(W1b, xb, y1b, nullptr, sum1, sq1);
    bn_finalize_kernel<<<1, 256, 0, stream>>>(sum1, sq1, gamma1, beta1, sc1, sh1, 1.f / (float)NTOT);
    bn_apply_bf16_kernel<<<(size_t)NTOT * CH / 8 / 256, 256, 0, stream>>>(y1b, sc1, sh1);

    gemm_kernel<CH, true><<<BATCH * (N1 / 64), 256, 0, stream>>>(W2b, y1b, nullptr, out, sum2, sq2);
    bn_finalize_kernel<<<1, 256, 0, stream>>>(sum2, sq2, gamma2, beta2, sc2, sh2, 1.f / (float)NTOT);
    bn_apply_f32_kernel<<<(size_t)NTOT * COUT / 4 / 256, 256, 0, stream>>>(out, sc2, sh2);
}
